// attention_fx_23854248362451
// MI455X (gfx1250) — matrix operand order repaired and checked by reading, not compiled, not run on hardware
//
#include <hip/hip_runtime.h>
#include <hip/hip_bf16.h>

typedef __attribute__((ext_vector_type(16))) __bf16 v16bf;
typedef __attribute__((ext_vector_type(8)))  float  v8f;

// ---------------------------------------------------------------------------
// WMMA helpers (CDNA5 v_wmma_f32_16x16x32_bf16, wave32; A/B use contiguous k = 16*(lane/16)+e, C per ISA 7.12.2)
// ---------------------------------------------------------------------------

__device__ __forceinline__ v8f wmma_bf16(v16bf a, v16bf b, v8f c) {
  return __builtin_amdgcn_wmma_f32_16x16x32_bf16(
      /*neg_a=*/false, a, /*neg_b=*/false, b,
      /*c_mod=*/(short)0, c, /*reuse_a=*/false, /*reuse_b=*/false);
}

// A-matrix 16x32 (MxK) from row-major f32 [16 rows, stride lda], K window at `a`.
// Lane L<16: row=L, K {0..15}; lane>=16: row=L-16, K {16..31}.
__device__ __forceinline__ v16bf load_a_frag(const float* __restrict__ a, int lda, int lane) {
  int r  = lane & 15;
  int g0 = (lane >> 4) * 16;
  int g1 = 8 + g0;
  const float* p = a + (size_t)r * lda;
  v16bf v;
#pragma unroll
  for (int e = 0; e < 8; ++e) v[e] = (__bf16)p[g0 + e];
#pragma unroll
  for (int e = 0; e < 8; ++e) v[8 + e] = (__bf16)p[g1 + e];
  return v;
}

// B-matrix 32x16 (KxN) where element(k,c) = b[c*lds + k] (K contiguous per lane).
__device__ __forceinline__ v16bf load_bt_frag(const float* __restrict__ b, int lds, int lane) {
  int c  = lane & 15;
  int kb = (lane >> 4) * 16;
  const float* p = b + (size_t)c * lds + kb;
  v16bf v;
#pragma unroll
  for (int e = 0; e < 16; ++e) v[e] = (__bf16)p[e];
  return v;
}

// C/D 16x16 f32: lane L: col = L&15; element e: row = e + (L>>4)*8.
__device__ __forceinline__ void store_c(float* __restrict__ out, int ldc, v8f c, int lane) {
  int col = lane & 15;
  int rb  = (lane >> 4) * 8;
#pragma unroll
  for (int e = 0; e < 8; ++e) out[(size_t)(rb + e) * ldc + col] = c[e];
}

// ---------------------------------------------------------------------------
// Pack a row-major f32 matrix [batches][R][C] into bf16 B-fragments:
// frag index ((b*(R/32)+kt)*(C/16)+ct), per-lane 16 contiguous bf16 (32 B).
// ---------------------------------------------------------------------------
__global__ __launch_bounds__(32) void pack_bfrag_kernel(
    const float* __restrict__ S, __bf16* __restrict__ F, int R, int C) {
  int lane = threadIdx.x;
  int ctiles = C >> 4, ktiles = R >> 5;
  int id = blockIdx.x;
  int ct = id % ctiles; id /= ctiles;
  int kt = id % ktiles; int b = id / ktiles;
  const float* Sb = S + (size_t)b * R * C;
  int c = lane & 15, kb = (lane >> 4) * 16;
  v16bf v;
#pragma unroll
  for (int e = 0; e < 16; ++e)
    v[e] = (__bf16)Sb[(size_t)(kt * 32 + kb + e) * C + ct * 16 + c];
  ((v16bf*)F)[(((size_t)b * ktiles + kt) * ctiles + ct) * 32 + lane] = v;
}

// ---------------------------------------------------------------------------
// Strip GEMM: Y[b] = A[b] @ Bmat, A f32 [batches][rows][K] row-major,
// Bmat pre-packed bf16 fragments (bstrideF = frags per batch, 0 => shared).
// One wave computes a 16 x (16*CT) strip: A-frag reused by CT WMMAs / K-step.
// ---------------------------------------------------------------------------
template <int CT>
__global__ __launch_bounds__(32) void gemm_af32_bpack_kernel(
    const float* __restrict__ A, const __bf16* __restrict__ BF, float* __restrict__ Y,
    int rows, int K, int cols, int bstrideF) {
  int lane = threadIdx.x;
  int groups = cols / (16 * CT), tilesR = rows >> 4;
  int id = blockIdx.x;
  int cg = id % groups; id /= groups;
  int rt = id % tilesR; int b = id / tilesR;
  int ctiles = cols >> 4;

  const float* Arow = A + ((size_t)b * rows + (size_t)rt * 16) * K;
  const v16bf* BFv = (const v16bf*)BF + (size_t)b * bstrideF * 32;
  v8f acc[CT];
#pragma unroll
  for (int t = 0; t < CT; ++t) acc[t] = (v8f){};

  for (int k = 0; k < K; k += 32) {
    v16bf a = load_a_frag(Arow + k, K, lane);
    int kt = k >> 5;
#pragma unroll
    for (int t = 0; t < CT; ++t) {
      v16bf bb = BFv[((size_t)kt * ctiles + cg * CT + t) * 32 + lane];
      acc[t] = wmma_bf16(a, bb, acc[t]);
    }
  }
#pragma unroll
  for (int t = 0; t < CT; ++t)
    store_c(Y + ((size_t)b * rows + (size_t)rt * 16) * cols + (cg * CT + t) * 16,
            cols, acc[t], lane);
}

// ---------------------------------------------------------------------------
// concat + minus scores (fused, tanh-bound VALU kernel).
// sjt_c[b,m,n] = sum_h tanh(s1[b,n,h]+s2[b,m,h])*vc[h]
// sjt_m[b,m,n] = sum_h tanh(sx[b,n,h]-sy[b,m,h])*vm[h]
// ---------------------------------------------------------------------------
__global__ __launch_bounds__(256) void cm_scores_kernel(
    const float* __restrict__ s1, const float* __restrict__ s2, const float* __restrict__ vc,
    const float* __restrict__ sx, const float* __restrict__ sy, const float* __restrict__ vm,
    float* __restrict__ sjt_c, float* __restrict__ sjt_m, int N, int M, int H) {
  __shared__ float l_s2[128], l_sy[128], l_vc[128], l_vm[128];
  int m = blockIdx.x % M;
  int b = blockIdx.x / M;
  int t = threadIdx.x;
  if (t < H) {
    size_t ro = ((size_t)b * M + m) * H + t;
    l_s2[t] = s2[ro];
    l_sy[t] = sy[ro];
    l_vc[t] = vc[t];
    l_vm[t] = vm[t];
  }
  __syncthreads();
  int n = t;
  const float4* s1r = (const float4*)(s1 + ((size_t)b * N + n) * H);
  const float4* sxr = (const float4*)(sx + ((size_t)b * N + n) * H);
  float ac = 0.f, am = 0.f;
#pragma unroll 2
  for (int h4 = 0; h4 < H / 4; ++h4) {
    float4 a4 = s1r[h4];
    float4 x4 = sxr[h4];
    int h = h4 * 4;
    ac += tanhf(a4.x + l_s2[h + 0]) * l_vc[h + 0];
    ac += tanhf(a4.y + l_s2[h + 1]) * l_vc[h + 1];
    ac += tanhf(a4.z + l_s2[h + 2]) * l_vc[h + 2];
    ac += tanhf(a4.w + l_s2[h + 3]) * l_vc[h + 3];
    am += tanhf(x4.x - l_sy[h + 0]) * l_vm[h + 0];
    am += tanhf(x4.y - l_sy[h + 1]) * l_vm[h + 1];
    am += tanhf(x4.z - l_sy[h + 2]) * l_vm[h + 2];
    am += tanhf(x4.w - l_sy[h + 3]) * l_vm[h + 3];
  }
  size_t o = ((size_t)b * M + m) * N + n;
  sjt_c[o] = ac;
  sjt_m[o] = am;
}

// ---------------------------------------------------------------------------
// bilinear scores: sjt[b,m,n] = sum_d y[b,m,d]*xb[b,n,d]
// ---------------------------------------------------------------------------
__global__ __launch_bounds__(32) void bilinear_scores_kernel(
    const float* __restrict__ Ym, const float* __restrict__ Xb, float* __restrict__ sjt,
    int M, int N, int D) {
  int lane = threadIdx.x;
  int tilesN = N >> 4, tilesM = M >> 4;
  int id = blockIdx.x;
  int nt = id % tilesN; id /= tilesN;
  int mt = id % tilesM; int b = id / tilesM;

  const float* Arow = Ym + ((size_t)b * M + (size_t)mt * 16) * D;
  const float* Bcol = Xb + ((size_t)b * N + (size_t)nt * 16) * D;
  v8f acc = {};
  for (int k = 0; k < D; k += 32) {
    v16bf a  = load_a_frag(Arow + k, D, lane);
    v16bf bb = load_bt_frag(Bcol + k, D, lane);
    acc = wmma_bf16(a, bb, acc);
  }
  store_c(sjt + ((size_t)b * M + (size_t)mt * 16) * N + nt * 16, N, acc, lane);
}

// ---------------------------------------------------------------------------
// dot-att scores (heavy kernel, ~17 GFLOP per call).
// For fixed (b,m): S = tanh((X[b] * Y[b,m]) @ Wd); sjt[b,m,n] = S[n,:].vd
// Workgroup = 8 waves over (b, n-tile, m-group-of-64): the 16xD X tile is
// staged once in LDS and reused by all 64 m-values handled by the block.
// Wave w handles m = mg*64 + w*8 + (0..7); 512 WMMAs per wave.
// ---------------------------------------------------------------------------
__global__ __launch_bounds__(256) void dot_scores_wg_kernel(
    const float* __restrict__ X, const float* __restrict__ Y,
    const __bf16* __restrict__ WdF, const float* __restrict__ vd,
    float* __restrict__ sjt, int Nn, int Mm, int D, int H) {
  __shared__ float lX[16 * 256];   // 16 rows x D f32 (D == 256)
  __shared__ float lvd[128];
  __shared__ float red[8][512];    // per-wave [16 rows][32 lanes]
  int tid  = threadIdx.x;
  int lane = tid & 31;
  int w    = tid >> 5;
  int ntiles  = Nn >> 4;
  int mgroups = Mm >> 6;
  int id = blockIdx.x;
  int mg = id % mgroups; id /= mgroups;
  int nt = id % ntiles;  int b = id / ntiles;
  int htiles = H >> 4;

  // stage X tile (16 rows x D, contiguous 16KB) + vd
  {
    const float4* src4 = (const float4*)(X + ((size_t)b * Nn + (size_t)nt * 16) * D);
    float4* dst4 = (float4*)lX;
    for (int i = tid; i < (16 * 256) / 4; i += 256) dst4[i] = src4[i];
    if (tid < H) lvd[tid] = vd[tid];
  }
  __syncthreads();

  const v16bf* WdFv = (const v16bf*)WdF;
  int r   = lane & 15;
  int g0  = (lane >> 4) * 16;
  int g1  = 8 + g0;
  int col = lane & 15;
  int rb  = (lane >> 4) * 8;

  for (int mi = 0; mi < 8; ++mi) {
    int m = mg * 64 + w * 8 + mi;
    const float* yrow = Y + ((size_t)b * Mm + m) * D;
    if (mi < 7) __builtin_prefetch(yrow + D, 0, 1);  // next m's y row

    v8f acc[8];
#pragma unroll
    for (int t = 0; t < 8; ++t) acc[t] = (v8f){};

    for (int k = 0; k < D; k += 32) {
      // A-fragment = (x row from LDS) * (y row from global), cvt to bf16
      const float4* x4a = (const float4*)(lX + r * 256 + k + g0);
      const float4* x4b = (const float4*)(lX + r * 256 + k + g1);
      const float4* y4a = (const float4*)(yrow + k + g0);
      const float4* y4b = (const float4*)(yrow + k + g1);
      float4 xa0 = x4a[0], xa1 = x4a[1], xb0 = x4b[0], xb1 = x4b[1];
      float4 ya0 = y4a[0], ya1 = y4a[1], yb0 = y4b[0], yb1 = y4b[1];
      v16bf a;
      a[0]  = (__bf16)(xa0.x * ya0.x); a[1]  = (__bf16)(xa0.y * ya0.y);
      a[2]  = (__bf16)(xa0.z * ya0.z); a[3]  = (__bf16)(xa0.w * ya0.w);
      a[4]  = (__bf16)(xa1.x * ya1.x); a[5]  = (__bf16)(xa1.y * ya1.y);
      a[6]  = (__bf16)(xa1.z * ya1.z); a[7]  = (__bf16)(xa1.w * ya1.w);
      a[8]  = (__bf16)(xb0.x * yb0.x); a[9]  = (__bf16)(xb0.y * yb0.y);
      a[10] = (__bf16)(xb0.z * yb0.z); a[11] = (__bf16)(xb0.w * yb0.w);
      a[12] = (__bf16)(xb1.x * yb1.x); a[13] = (__bf16)(xb1.y * yb1.y);
      a[14] = (__bf16)(xb1.z * yb1.z); a[15] = (__bf16)(xb1.w * yb1.w);

      int kt = k >> 5;
#pragma unroll
      for (int t = 0; t < 8; ++t) {
        v16bf bb = WdFv[((size_t)kt * htiles + t) * 32 + lane];
        acc[t] = wmma_bf16(a, bb, acc[t]);
      }
    }

    // tanh, dot with vd, cross-lane reduce through per-wave LDS slab.
    // Rows 0-7 live only in lanes 0-15, rows 8-15 only in lanes 16-31.
#pragma unroll
    for (int e = 0; e < 8; ++e) {
      float s = 0.f;
#pragma unroll
      for (int t = 0; t < 8; ++t) s += tanhf(acc[t][e]) * lvd[t * 16 + col];
      red[w][(rb + e) * 32 + lane] = s;
    }
    __syncthreads();  // uniform trip count across all 8 waves
    if (lane < 16) {
      int lb = (lane < 8) ? 0 : 16;
      float s = 0.f;
#pragma unroll
      for (int j = 0; j < 16; ++j) s += red[w][lane * 32 + lb + j];
      sjt[((size_t)b * Mm + m) * Nn + nt * 16 + lane] = s;
    }
    __syncthreads();
  }
}

// ---------------------------------------------------------------------------
// row softmax in place (N == 256 == blockDim)
// ---------------------------------------------------------------------------
__global__ __launch_bounds__(256) void softmax_rows_kernel(float* __restrict__ sjt, int N) {
  __shared__ float red[256];
  size_t row = blockIdx.x;
  int t = threadIdx.x;
  float v = sjt[row * N + t];
  red[t] = v;
  __syncthreads();
  for (int s = 128; s > 0; s >>= 1) {
    if (t < s) red[t] = fmaxf(red[t], red[t + s]);
    __syncthreads();
  }
  float mx = red[0];
  __syncthreads();
  float e = __expf(v - mx);
  red[t] = e;
  __syncthreads();
  for (int s = 128; s > 0; s >>= 1) {
    if (t < s) red[t] += red[t + s];
    __syncthreads();
  }
  sjt[row * N + t] = e / red[0];
}

// ---------------------------------------------------------------------------
// elementwise max of 6 tensors
// ---------------------------------------------------------------------------
__global__ __launch_bounds__(256) void final_max_kernel(
    const float* __restrict__ a0, const float* __restrict__ a1, const float* __restrict__ a2,
    const float* __restrict__ a3, const float* __restrict__ a4, const float* __restrict__ a5,
    float* __restrict__ out, int n) {
  int i = blockIdx.x * 256 + threadIdx.x;
  if (i < n) {
    float v = a0[i];
    v = fmaxf(v, a1[i]);
    v = fmaxf(v, a2[i]);
    v = fmaxf(v, a3[i]);
    v = fmaxf(v, a4[i]);
    v = fmaxf(v, a5[i]);
    out[i] = v;
  }
}

// ---------------------------------------------------------------------------
extern "C" void kernel_launch(void* const* d_in, const int* in_sizes, int n_in,
                              void* d_out, int out_size, void* d_ws, size_t ws_size,
                              hipStream_t stream) {
  (void)in_sizes; (void)n_in; (void)out_size; (void)ws_size;
  const int B = 4, N = 256, M = 256, D = 256, H = 128;

  const float* x0  = (const float*)d_in[0];
  const float* x1  = (const float*)d_in[1];
  const float* Wc1 = (const float*)d_in[2];
  const float* Wc2 = (const float*)d_in[3];
  const float* vc  = (const float*)d_in[4];
  const float* Wb  = (const float*)d_in[5];
  const float* Wd1 = (const float*)d_in[6];
  const float* vd1 = (const float*)d_in[7];
  const float* Wd2 = (const float*)d_in[8];
  const float* vd2 = (const float*)d_in[9];
  const float* Wm  = (const float*)d_in[10];
  const float* vm  = (const float*)d_in[11];
  float* out = (float*)d_out;

  // ---- f32 workspace regions ----
  float* ws = (float*)d_ws;
  size_t o = 0;
  float* s1 = ws + o; o += (size_t)B * N * H;
  float* s2 = ws + o; o += (size_t)B * M * H;
  float* sx = ws + o; o += (size_t)B * N * H;
  float* sy = ws + o; o += (size_t)B * M * H;
  float* xb = ws + o; o += (size_t)B * N * D;
  float* sjt_c  = ws + o; o += (size_t)B * M * N;
  float* sjt_b  = ws + o; o += (size_t)B * M * N;
  float* sjt_d1 = ws + o; o += (size_t)B * M * N;
  float* sjt_d2 = ws + o; o += (size_t)B * M * N;
  float* sjt_m  = ws + o; o += (size_t)B * M * N;
  float* qtc = ws + o; o += (size_t)B * M * D;
  float* qtb = ws + o; o += (size_t)B * M * D;
  float* qts = ws + o; o += (size_t)B * M * D;
  float* qtd = ws + o; o += (size_t)B * M * D;
  float* qtm = ws + o; o += (size_t)B * M * D;

  // ---- bf16 pre-packed B-fragment regions (each size multiple of 512) ----
  const size_t fDH = (size_t)(D / 32) * (H / 16) * 512;      // 256x128 weight
  const size_t fDD = (size_t)(D / 32) * (D / 16) * 512;      // 256x256 weight
  const size_t fV  = (size_t)B * (N / 32) * (D / 16) * 512;  // batched values
  __bf16* bfws = (__bf16*)(ws + o);
  size_t ob = 0;
  __bf16* Wc1F = bfws + ob; ob += fDH;
  __bf16* Wc2F = bfws + ob; ob += fDH;
  __bf16* WmF  = bfws + ob; ob += fDH;
  __bf16* Wd1F = bfws + ob; ob += fDH;
  __bf16* Wd2F = bfws + ob; ob += fDH;
  __bf16* WbF  = bfws + ob; ob += fDD;
  __bf16* x0F  = bfws + ob; ob += fV;
  __bf16* x1F  = bfws + ob; ob += fV;

  // --- stage 0: pack B operands into WMMA fragment layout (bf16) ---
  int gDH = (D / 32) * (H / 16);           // 64
  int gDD = (D / 32) * (D / 16);           // 128
  int gV  = B * (N / 32) * (D / 16);       // 512
  pack_bfrag_kernel<<<gDH, 32, 0, stream>>>(Wc1, Wc1F, D, H);
  pack_bfrag_kernel<<<gDH, 32, 0, stream>>>(Wc2, Wc2F, D, H);
  pack_bfrag_kernel<<<gDH, 32, 0, stream>>>(Wm,  WmF,  D, H);
  pack_bfrag_kernel<<<gDH, 32, 0, stream>>>(Wd1, Wd1F, D, H);
  pack_bfrag_kernel<<<gDH, 32, 0, stream>>>(Wd2, Wd2F, D, H);
  pack_bfrag_kernel<<<gDD, 32, 0, stream>>>(Wb,  WbF,  D, D);
  pack_bfrag_kernel<<<gV,  32, 0, stream>>>(x0,  x0F,  N, D);
  pack_bfrag_kernel<<<gV,  32, 0, stream>>>(x1,  x1F,  N, D);

  // --- stage 1: input GEMMs (strip CT=8, shared weights => bstrideF 0) ---
  int g_h = B * (N / 16) * (H / 128);      // 64 waves
  gemm_af32_bpack_kernel<8><<<g_h, 32, 0, stream>>>(x0, Wc1F, s1, N, D, H, 0);
  gemm_af32_bpack_kernel<8><<<g_h, 32, 0, stream>>>(x1, Wc2F, s2, M, D, H, 0);
  gemm_af32_bpack_kernel<8><<<g_h, 32, 0, stream>>>(x0, WmF,  sx, N, D, H, 0);
  gemm_af32_bpack_kernel<8><<<g_h, 32, 0, stream>>>(x1, WmF,  sy, M, D, H, 0);
  int g_d = B * (N / 16) * (D / 128);      // 128 waves
  gemm_af32_bpack_kernel<8><<<g_d, 32, 0, stream>>>(x0, WbF, xb, N, D, D, 0);

  // --- stage 2: scores ---
  cm_scores_kernel<<<B * M, 256, 0, stream>>>(s1, s2, vc, sx, sy, vm, sjt_c, sjt_m, N, M, H);
  bilinear_scores_kernel<<<B * (M / 16) * (N / 16), 32, 0, stream>>>(x1, xb, sjt_b, M, N, D);
  int g_dot = B * (N / 16) * (M / 64);     // 256 blocks x 8 waves
  dot_scores_wg_kernel<<<g_dot, 256, 0, stream>>>(x0, x1, Wd1F, vd1, sjt_d1, N, M, D, H);
  dot_scores_wg_kernel<<<g_dot, 256, 0, stream>>>(x1, x0, Wd2F, vd2, sjt_d2, M, N, D, H);

  // --- stage 3: softmax over n (in place) ---
  softmax_rows_kernel<<<B * M, 256, 0, stream>>>(sjt_c,  N);
  softmax_rows_kernel<<<B * M, 256, 0, stream>>>(sjt_b,  N);
  softmax_rows_kernel<<<B * M, 256, 0, stream>>>(sjt_d1, N);
  softmax_rows_kernel<<<B * M, 256, 0, stream>>>(sjt_d2, N);
  softmax_rows_kernel<<<B * M, 256, 0, stream>>>(sjt_m,  N);

  // --- stage 4: probability @ values (same strip kernel, batched packed V) ---
  int bstrV = (N / 32) * (D / 16);         // fragments per batch of V
  int g_p = B * (M / 16) * (D / 128);      // 128 waves each
  gemm_af32_bpack_kernel<8><<<g_p, 32, 0, stream>>>(sjt_c,  x0F, qtc, M, N, D, bstrV);
  gemm_af32_bpack_kernel<8><<<g_p, 32, 0, stream>>>(sjt_b,  x0F, qtb, M, N, D, bstrV);
  gemm_af32_bpack_kernel<8><<<g_p, 32, 0, stream>>>(sjt_d1, x0F, qts, M, N, D, bstrV);
  gemm_af32_bpack_kernel<8><<<g_p, 32, 0, stream>>>(sjt_d2, x1F, qtd, M, N, D, bstrV);
  gemm_af32_bpack_kernel<8><<<g_p, 32, 0, stream>>>(sjt_m,  x0F, qtm, M, N, D, bstrV);

  // --- stage 5: elementwise max ---
  int n = B * M * D;
  final_max_kernel<<<(n + 255) / 256, 256, 0, stream>>>(x1, qts, qtc, qtd, qtb, qtm, out, n);
}
